// MaskingGCN_60181081752120
// MI455X (gfx1250) — compile-verified, matrix operand order checked by reading, not run on hardware
//
#include <hip/hip_runtime.h>
#include <hip/hip_bf16.h>

typedef __attribute__((ext_vector_type(16))) __bf16 v16bf;
typedef __attribute__((ext_vector_type(8)))  __bf16 v8bf;
typedef __attribute__((ext_vector_type(8)))  float  v8f;

#define HDIM 256

// ---------------- utility kernels ----------------

__global__ void zero_f32_k(float* __restrict__ p, long n4) {
  long i = (long)blockIdx.x * blockDim.x + threadIdx.x;
  if (i < n4) ((float4*)p)[i] = make_float4(0.f, 0.f, 0.f, 0.f);
}

__global__ void count_deg_k(const int* __restrict__ dst, int E, float* __restrict__ deg) {
  int e = blockIdx.x * blockDim.x + threadIdx.x;
  if (e < E) atomicAdd(&deg[dst[e]], 1.0f);
}

__global__ void inv_deg_k(float* __restrict__ deg, int N) {
  int i = blockIdx.x * blockDim.x + threadIdx.x;
  if (i < N) deg[i] = 1.0f / fmaxf(deg[i], 1.0f);
}

// W: [256 in][256 out] f32 row-major  ->  WT: [256 out][256 in] bf16 row-major
__global__ void transpose_w_k(const float* __restrict__ w, __bf16* __restrict__ wt) {
  int i = blockIdx.x * 16 + threadIdx.x;  // in
  int o = blockIdx.y * 16 + threadIdx.y;  // out
  wt[o * HDIM + i] = (__bf16)w[i * HDIM + o];
}

// One wave32 per edge: agg[dst] += h[src]  (L2-resident scatter; 8 atomics/lane)
__global__ void scatter_add_k(const float* __restrict__ h, const int* __restrict__ src,
                              const int* __restrict__ dst, int E, float* __restrict__ agg) {
  int wave = (int)((blockIdx.x * (long)blockDim.x + threadIdx.x) >> 5);
  int lane = threadIdx.x & 31;
  if (wave >= E) return;
  int s = src[wave];
  int d = dst[wave];
  const float* hp = h + (long)s * HDIM;
  float* ap = agg + (long)d * HDIM;
#pragma unroll
  for (int j = 0; j < 8; ++j)
    atomicAdd(&ap[lane + 32 * j], hp[lane + 32 * j]);
}

// h16 = bf16(h) ; a16 = bf16(agg * inv_deg[row])  (4 elems/thread)
__global__ void convert_bf16_k(const float* __restrict__ h, const float* __restrict__ agg,
                               const float* __restrict__ invd, long total4,
                               __bf16* __restrict__ h16, __bf16* __restrict__ a16) {
  long t = (long)blockIdx.x * blockDim.x + threadIdx.x;
  if (t >= total4) return;
  long i = t * 4;
  int row = (int)(i >> 8);
  float s = invd[row];
  float4 hv = ((const float4*)h)[t];
  float4 av = ((const float4*)agg)[t];
  struct bf4 { __bf16 x, y, z, w; };
  ((bf4*)h16)[t] = bf4{(__bf16)hv.x, (__bf16)hv.y, (__bf16)hv.z, (__bf16)hv.w};
  ((bf4*)a16)[t] = bf4{(__bf16)(av.x * s), (__bf16)(av.y * s),
                       (__bf16)(av.z * s), (__bf16)(av.w * s)};
}

// ---------------- WMMA GEMM ----------------
// Fragment loader for 16-bit A/B (16x32 K-tile) per ISA 7.12.2:
// lanes 0-15  : row (l&15), K = k0+{0..7}   and k0+16+{0..7}
// lanes 16-31 : row (l&15), K = k0+{8..15}  and k0+16+{8..15}
__device__ __forceinline__ v16bf load_frag(const __bf16* __restrict__ base, int row0,
                                           int k0, int lane, int maxRow) {
  int r = row0 + (lane & 15);
  if (r > maxRow) r = maxRow;
  int ko = k0 + ((lane >> 4) << 3);
  const __bf16* p = base + (long)r * HDIM + ko;
  union { v16bf v; v8bf h[2]; } u;
  u.h[0] = *(const v8bf*)(p);
  u.h[1] = *(const v8bf*)(p + 16);
  return u.v;
}

__device__ __forceinline__ v16bf wmma_bf16(v16bf a, v16bf b, v8f& acc) {
  acc = __builtin_amdgcn_wmma_f32_16x16x32_bf16(false, a, false, b,
                                                (short)0, acc, false, false);
  return a;
}

// out[N,256] = act( a16 @ WAt^T + h16 @ WBt^T )
// Register-blocked: each wave owns a 32x32 output tile (2 m-tiles x 2 n-tiles,
// 4 f32 accumulators). Per K-step: 8 fragment loads feed 8 WMMAs (1 load/wmma).
// Block = 8 waves = one 32-row m-strip covering all 8 n-pairs (A hot in L0).
__global__ void __launch_bounds__(256)
gcn_layer_k(const __bf16* __restrict__ a16, const __bf16* __restrict__ h16,
            const __bf16* __restrict__ WAt, const __bf16* __restrict__ WBt,
            float* __restrict__ out, int N, int applySigmoid) {
  int lane = threadIdx.x & 31;
  int m0 = blockIdx.x * 32;          // m-pair base for whole block
  if (m0 >= N) return;               // wave-uniform: EXEC all-ones for WMMA
  int m1 = m0 + 16;
  int n0 = (threadIdx.x >> 5) * 32;  // n-pair base for this wave
  int n1 = n0 + 16;
  int maxRow = N - 1;

  v8f acc00 = {}, acc01 = {}, acc10 = {}, acc11 = {};
#pragma unroll
  for (int k0 = 0; k0 < HDIM; k0 += 32) {
    // A-side fragments (rows of agg / h)
    v16bf fa0 = load_frag(a16, m0, k0, lane, maxRow);
    v16bf fa1 = load_frag(a16, m1, k0, lane, maxRow);
    v16bf fh0 = load_frag(h16, m0, k0, lane, maxRow);
    v16bf fh1 = load_frag(h16, m1, k0, lane, maxRow);
    // B-side fragments (columns of the two weight matrices, from transposed copies)
    v16bf wa0 = load_frag(WAt, n0, k0, lane, HDIM - 1);
    v16bf wa1 = load_frag(WAt, n1, k0, lane, HDIM - 1);
    v16bf wb0 = load_frag(WBt, n0, k0, lane, HDIM - 1);
    v16bf wb1 = load_frag(WBt, n1, k0, lane, HDIM - 1);

    wmma_bf16(fa0, wa0, acc00);
    wmma_bf16(fh0, wb0, acc00);
    wmma_bf16(fa0, wa1, acc01);
    wmma_bf16(fh0, wb1, acc01);
    wmma_bf16(fa1, wa0, acc10);
    wmma_bf16(fh1, wb0, acc10);
    wmma_bf16(fa1, wa1, acc11);
    wmma_bf16(fh1, wb1, acc11);
  }

  // C/D layout: VGPR r -> row base + r + 8*(lane>=16), col nbase + (lane&15)
  int c0 = n0 + (lane & 15);
  int c1 = n1 + (lane & 15);
  int r0 = m0 + ((lane >> 4) << 3);
  int r1 = m1 + ((lane >> 4) << 3);
#pragma unroll
  for (int r = 0; r < 8; ++r) {
    float v00 = acc00[r], v01 = acc01[r], v10 = acc10[r], v11 = acc11[r];
    if (applySigmoid) {
      v00 = 1.0f / (1.0f + __expf(-v00));
      v01 = 1.0f / (1.0f + __expf(-v01));
      v10 = 1.0f / (1.0f + __expf(-v10));
      v11 = 1.0f / (1.0f + __expf(-v11));
    }
    int row0 = r0 + r, row1 = r1 + r;
    if (row0 < N) {
      out[(long)row0 * HDIM + c0] = v00;
      out[(long)row0 * HDIM + c1] = v01;
    }
    if (row1 < N) {
      out[(long)row1 * HDIM + c0] = v10;
      out[(long)row1 * HDIM + c1] = v11;
    }
  }
}

// out[n] = dot(h[n,:], W_out) + b ; one wave per node, shuffle reduction
__global__ void proj_out_k(const float* __restrict__ h, const float* __restrict__ w,
                           const float* __restrict__ b, float* __restrict__ out, int N) {
  int wave = (int)((blockIdx.x * (long)blockDim.x + threadIdx.x) >> 5);
  int lane = threadIdx.x & 31;
  if (wave >= N) return;
  const float* hp = h + (long)wave * HDIM;
  float s = 0.f;
#pragma unroll
  for (int j = 0; j < 8; ++j) s += hp[lane + 32 * j] * w[lane + 32 * j];
#pragma unroll
  for (int off = 16; off > 0; off >>= 1) s += __shfl_xor(s, off, 32);
  if (lane == 0) out[wave] = s + b[0];
}

// ---------------- launch ----------------

extern "C" void kernel_launch(void* const* d_in, const int* in_sizes, int n_in,
                              void* d_out, int out_size, void* d_ws, size_t ws_size,
                              hipStream_t stream) {
  const float* x       = (const float*)d_in[0];
  const int*   ei      = (const int*)  d_in[1];
  const float* WA_in   = (const float*)d_in[2];
  const float* WB_in   = (const float*)d_in[3];
  const float* A_stack = (const float*)d_in[4];
  const float* B_stack = (const float*)d_in[5];
  const float* W_out   = (const float*)d_in[6];
  const float* b_out   = (const float*)d_in[7];

  const int N = in_sizes[0] / HDIM;
  const int E = in_sizes[1] / 2;
  const int L = in_sizes[4] / (HDIM * HDIM);
  const int* src = ei;
  const int* dst = ei + E;

  // workspace carve-up (16B-aligned slabs)
  char* ws = (char*)d_ws;
  float*  bufA = (float*)ws;  ws += (size_t)N * HDIM * 4;
  float*  bufB = (float*)ws;  ws += (size_t)N * HDIM * 4;
  __bf16* h16  = (__bf16*)ws; ws += (size_t)N * HDIM * 2;
  __bf16* a16  = (__bf16*)ws; ws += (size_t)N * HDIM * 2;
  float*  invd = (float*)ws;  ws += ((size_t)N * 4 + 15) & ~(size_t)15;
  __bf16* wt[6];
  for (int i = 0; i < 6; ++i) { wt[i] = (__bf16*)ws; ws += (size_t)HDIM * HDIM * 2; }

  const long feat  = (long)N * HDIM;
  const long feat4 = feat / 4;

  // degree -> inv_deg
  zero_f32_k<<<(N / 4 + 256) / 256, 256, 0, stream>>>(invd, (N + 3) / 4);
  count_deg_k<<<(E + 255) / 256, 256, 0, stream>>>(dst, E, invd);
  inv_deg_k<<<(N + 255) / 256, 256, 0, stream>>>(invd, N);

  // pre-transpose + bf16-convert all 6 weight matrices
  transpose_w_k<<<dim3(16, 16), dim3(16, 16), 0, stream>>>(WA_in, wt[0]);
  transpose_w_k<<<dim3(16, 16), dim3(16, 16), 0, stream>>>(WB_in, wt[1]);
  for (int l = 0; l < L; ++l) {
    transpose_w_k<<<dim3(16, 16), dim3(16, 16), 0, stream>>>(A_stack + (size_t)l * HDIM * HDIM, wt[2 + 2 * l]);
    transpose_w_k<<<dim3(16, 16), dim3(16, 16), 0, stream>>>(B_stack + (size_t)l * HDIM * HDIM, wt[3 + 2 * l]);
  }

  const int gemm_blocks = (N + 31) / 32;  // one 32-row m-strip per block

  const float* cur = x;
  for (int l = 0; l <= L; ++l) {
    float* ab;  // agg buffer
    float* ob;  // layer output (reuses agg buffer after conversion when possible)
    if (l == 0) { ab = bufA; ob = bufB; }
    else        { ab = (cur == bufA) ? bufB : bufA; ob = ab; }

    zero_f32_k<<<(int)((feat4 + 255) / 256), 256, 0, stream>>>(ab, feat4);
    scatter_add_k<<<(int)(((long)E * 32 + 255) / 256), 256, 0, stream>>>(cur, src, dst, E, ab);
    convert_bf16_k<<<(int)((feat4 + 255) / 256), 256, 0, stream>>>(cur, ab, invd, feat4, h16, a16);
    gcn_layer_k<<<gemm_blocks, 256, 0, stream>>>(a16, h16, wt[2 * l], wt[2 * l + 1],
                                                 ob, N, l > 0 ? 1 : 0);
    cur = ob;
  }

  proj_out_k<<<(int)(((long)N * 32 + 255) / 256), 256, 0, stream>>>(cur, W_out, b_out,
                                                                    (float*)d_out, N);
}
